// SpatialAttentionEnhanceBlock_5291399709095
// MI455X (gfx1250) — matrix operand order repaired and checked by reading, not compiled, not run on hardware
//
#include <hip/hip_runtime.h>
#include <hip/hip_bf16.h>
#include <math.h>

// Problem dims (fixed by reference)
#define BDIM 4
#define CDIM 128
#define NDIM 512
#define TDIM 16
#define HDIM 4
#define DKDIM 32
// 1/SCALE = 1/sqrt(32)
#define INV_SCALE 0.17677669529663687f

typedef __attribute__((ext_vector_type(16))) __bf16 v16bf;
typedef __attribute__((ext_vector_type(8)))  float v8f;

union BfVec { v16bf v; uint4 q[2]; };

// Load a lane's 16 bf16 WMMA operand elements as two aligned 16B chunks.
static __device__ inline v16bf load2x16B(const __hip_bfloat16* p0, const __hip_bfloat16* p1) {
  BfVec u;
  u.q[0] = *reinterpret_cast<const uint4*>(p0);
  u.q[1] = *reinterpret_cast<const uint4*>(p1);
  return u.v;
}

static __device__ inline float warpReduceMax(float v) {
#pragma unroll
  for (int off = 16; off > 0; off >>= 1) v = fmaxf(v, __shfl_xor(v, off, 32));
  return v;
}
static __device__ inline float warpReduceSum(float v) {
#pragma unroll
  for (int off = 16; off > 0; off >>= 1) v += __shfl_xor(v, off, 32);
  return v;
}
static __device__ inline float blockReduceMax(float v, float* red) {
  const int lane = threadIdx.x & 31, w = threadIdx.x >> 5, nw = (int)(blockDim.x >> 5);
  v = warpReduceMax(v);
  if (lane == 0) red[w] = v;
  __syncthreads();
  float r = red[0];
  for (int i = 1; i < nw; ++i) r = fmaxf(r, red[i]);
  __syncthreads();
  return r;
}
static __device__ inline float blockReduceSum(float v, float* red) {
  const int lane = threadIdx.x & 31, w = threadIdx.x >> 5, nw = (int)(blockDim.x >> 5);
  v = warpReduceSum(v);
  if (lane == 0) red[w] = v;
  __syncthreads();
  float r = 0.f;
  for (int i = 0; i < nw; ++i) r += red[i];
  __syncthreads();
  return r;
}

// ---------------------------------------------------------------------------
// Kernel 1: Q/K projection + bf16 transpose of x.
// grid = B*T*N blocks of 128 threads; thread index == output channel o == load channel c.
//   Qb/Kb layout: [b][t][n][o]  (o = h*32+d, so d contiguous per head)
//   xT  layout: [b][c][t][n]   (j=n contiguous -> B-operand rows for final GEMM)
// ---------------------------------------------------------------------------
__global__ void __launch_bounds__(128)
proj_qk_kernel(const float* __restrict__ x, const float* __restrict__ Wq,
               const float* __restrict__ Wk,
               __hip_bfloat16* __restrict__ Qb, __hip_bfloat16* __restrict__ Kb,
               __hip_bfloat16* __restrict__ xT) {
  __shared__ float xs[CDIM];
  const int bx  = blockIdx.x;
  const int b   = bx >> 13;          // T*N = 8192 per batch
  const int rem = bx & 8191;
  const int t   = rem >> 9;
  const int n   = rem & 511;
  const int o   = threadIdx.x;       // 0..127

  const float xv = x[(((size_t)b * CDIM + o) * NDIM + n) * TDIM + t];
  xs[o] = xv;
  __syncthreads();

  const float* wq = Wq + (size_t)o * CDIM;
  const float* wk = Wk + (size_t)o * CDIM;
  float q = 0.f, k = 0.f;
#pragma unroll 8
  for (int c = 0; c < CDIM; ++c) {
    const float xc = xs[c];
    q = fmaf(wq[c], xc, q);
    k = fmaf(wk[c], xc, k);
  }
  const size_t qidx = (((size_t)b * TDIM + t) * NDIM + n) * CDIM + o;
  Qb[qidx] = __float2bfloat16(q);
  Kb[qidx] = __float2bfloat16(k);
  xT[(((size_t)b * CDIM + o) * TDIM + t) * NDIM + n] = __float2bfloat16(xv);
}

// ---------------------------------------------------------------------------
// Kernel 2: row softmax of structural -> att_struct [N][N]
// ---------------------------------------------------------------------------
__global__ void __launch_bounds__(256)
struct_softmax_kernel(const float* __restrict__ structural, float* __restrict__ att_struct) {
  __shared__ float red[8];
  const int i = blockIdx.x;
  const int tid = threadIdx.x;
  const float* row = structural + (size_t)i * NDIM;
  const float v0 = row[tid], v1 = row[tid + 256];
  const float m  = blockReduceMax(fmaxf(v0, v1), red);
  const float e0 = __expf(v0 - m), e1 = __expf(v1 - m);
  const float s  = blockReduceSum(e0 + e1, red);
  const float inv = 1.0f / s;
  float* dst = att_struct + (size_t)i * NDIM;
  dst[tid] = e0 * inv;
  dst[tid + 256] = e1 * inv;
}

// ---------------------------------------------------------------------------
// Kernel 3: scores (WMMA bf16) + fused softmax + mean over (t,h) -> att_node [B][N][N]
// grid = B * (N/16) blocks; 128 threads = 4 waves; 160KB dynamic LDS:
//   per-wave score scratch 16x512 f32 (4x32KB) + shared accumulator 16x512 f32 (32KB)
// ---------------------------------------------------------------------------
__global__ void __launch_bounds__(128)
scores_softmax_kernel(const __hip_bfloat16* __restrict__ Qb,
                      const __hip_bfloat16* __restrict__ Kb,
                      float* __restrict__ att_node) {
  extern __shared__ float smem[];
  const int lane = threadIdx.x & 31;
  const int wave = threadIdx.x >> 5;
  const int b  = blockIdx.x >> 5;
  const int n0 = (blockIdx.x & 31) << 4;

  float* scores = smem + (size_t)wave * (16 * NDIM);
  float* accum  = smem + (size_t)4    * (16 * NDIM);
  for (int i = threadIdx.x; i < 16 * NDIM; i += 128) accum[i] = 0.0f;
  __syncthreads();

  const int hiHalf = (lane >= 16);
  const int rr     = lane & 15;
  const int koffA  = hiHalf ? 8 : 0;   // A operand k-offset (16-bit A layout, ISA 7.12.2)
  const int koffB  = hiHalf ? 8 : 0;   // B operand k-offset

  for (int p = wave; p < TDIM * HDIM; p += 4) {   // 16 (t,h) pairs per wave
    const int t = p & (TDIM - 1);
    const int h = p >> 4;
    const size_t base = (((size_t)b * TDIM + t) * NDIM) * CDIM + (size_t)h * DKDIM;

    const __hip_bfloat16* qrow = Qb + base + (size_t)(n0 + rr) * CDIM;
    const v16bf aQ = load2x16B(qrow + koffA, qrow + 16 + koffA);

    // 32 column tiles: scores[16][512] for this (t,h)
    for (int mt = 0; mt < 32; ++mt) {
      const __hip_bfloat16* krow = Kb + base + (size_t)(mt * 16 + rr) * CDIM + koffB;
      const v16bf bK = load2x16B(krow, krow + 16);
      v8f c = {0.f, 0.f, 0.f, 0.f, 0.f, 0.f, 0.f, 0.f};
      c = __builtin_amdgcn_wmma_f32_16x16x32_bf16(false, aQ, false, bK, (short)0, c,
                                                  false, false);
      const int col = mt * 16 + rr;
      const int rb  = hiHalf ? 8 : 0;
#pragma unroll
      for (int r = 0; r < 8; ++r) scores[(rb + r) * NDIM + col] = c[r];
    }

    // per-row softmax over 512 cols, accumulate 1/64-scaled into shared accumulator
    for (int r = 0; r < 16; ++r) {
      float ev[16];
      float m = -INFINITY;
#pragma unroll
      for (int kk = 0; kk < 16; ++kk) {
        const float v = scores[r * NDIM + lane + 32 * kk] * INV_SCALE;
        ev[kk] = v;
        m = fmaxf(m, v);
      }
      m = warpReduceMax(m);
      float s = 0.f;
#pragma unroll
      for (int kk = 0; kk < 16; ++kk) { ev[kk] = __expf(ev[kk] - m); s += ev[kk]; }
      s = warpReduceSum(s);
      const float rinv = 1.0f / (s * (float)(TDIM * HDIM));  // softmax * mean(T*H)
#pragma unroll
      for (int kk = 0; kk < 16; ++kk)
        atomicAdd(&accum[r * NDIM + lane + 32 * kk], ev[kk] * rinv);
    }
  }
  __syncthreads();
  float* dst = att_node + ((size_t)b * NDIM + n0) * NDIM;
  for (int i = threadIdx.x; i < 16 * NDIM; i += 128) dst[i] = accum[i];
}

// ---------------------------------------------------------------------------
// Kernel 4: fuse + symmetrize + row softmax -> fused_bf [B][N][N] (bf16)
// sym = 0.5*(fw0*(an_ij+an_ji) + fw1*(as_ij+as_ji)) + fb
// ---------------------------------------------------------------------------
__global__ void __launch_bounds__(256)
fuse_softmax_kernel(const float* __restrict__ att_node, const float* __restrict__ att_struct,
                    const float* __restrict__ fusion_w, const float* __restrict__ fusion_b,
                    __hip_bfloat16* __restrict__ fused_bf) {
  __shared__ float red[8];
  const int b = blockIdx.x >> 9;
  const int i = blockIdx.x & 511;
  const int tid = threadIdx.x;
  const float fw0 = fusion_w[0], fw1 = fusion_w[1], fb = fusion_b[0];
  const float* anb = att_node + (size_t)b * NDIM * NDIM;

  float vals[2];
#pragma unroll
  for (int k = 0; k < 2; ++k) {
    const int j = tid + k * 256;
    const float an_ij = anb[(size_t)i * NDIM + j];
    const float an_ji = anb[(size_t)j * NDIM + i];
    const float as_ij = att_struct[(size_t)i * NDIM + j];
    const float as_ji = att_struct[(size_t)j * NDIM + i];
    vals[k] = 0.5f * (fw0 * (an_ij + an_ji) + fw1 * (as_ij + as_ji)) + fb;
  }
  const float m  = blockReduceMax(fmaxf(vals[0], vals[1]), red);
  const float e0 = __expf(vals[0] - m), e1 = __expf(vals[1] - m);
  const float s  = blockReduceSum(e0 + e1, red);
  const float inv = 1.0f / s;
  __hip_bfloat16* dst = fused_bf + ((size_t)b * NDIM + i) * NDIM;
  dst[tid]       = __float2bfloat16(e0 * inv);
  dst[tid + 256] = __float2bfloat16(e1 * inv);
}

// ---------------------------------------------------------------------------
// Kernel 5: out[b,c,n,t] = sum_j fused[b,n,j] * x[b,c,j,t]  via WMMA bf16
// Per b: GEMM M=512(n) x K=512(j) x 2048 cols(c*T+t).
// 4 waves/block share one async-staged xT B-panel (128 cols x 32 j, bf16) in LDS:
//   global_load_async_to_lds_b128 + s_wait_asynccnt (ASYNCcnt path), then
//   conflict-free ds_load_b128 reads (80B padded column stride).
// grid = B * 8(n-groups of 64 rows) * 16(col groups of 128); 128 threads.
// ---------------------------------------------------------------------------
#define AGG_COLS   128
#define AGG_STRIDE 80   // 32 bf16 = 64B payload + 16B pad -> 16 lanes hit 16 bank groups

__global__ void __launch_bounds__(128)
aggregate_kernel(const __hip_bfloat16* __restrict__ fused_bf,
                 const __hip_bfloat16* __restrict__ xT,
                 float* __restrict__ out) {
  __shared__ __align__(16) unsigned char xtile[AGG_COLS * AGG_STRIDE]; // 10 KB
  const int tid  = threadIdx.x;
  const int lane = tid & 31;
  const int wave = tid >> 5;
  const int bx  = blockIdx.x;
  const int b   = bx >> 7;                       // 8*16 = 128 blocks per batch
  const int rem = bx & 127;
  const int n0  = ((rem >> 4) << 6) + (wave << 4);  // n-group*64 + wave*16
  const int cg  = rem & 15;                      // column group (128 cols each)

  const int hiHalf = (lane >= 16);
  const int rr     = lane & 15;
  const int koffA  = hiHalf ? 8 : 0;
  const int koffB  = hiHalf ? 8 : 0;

  const __hip_bfloat16* arow = fused_bf + ((size_t)b * NDIM + n0 + rr) * NDIM;
  // this thread stages xT column (c*T+t) = cg*128 + tid
  const __hip_bfloat16* gcol =
      xT + ((size_t)b * (CDIM * TDIM) + (size_t)cg * AGG_COLS + tid) * NDIM;
  // LDS byte offset of this thread's staging column (generic ptr low 32 bits = LDS addr)
  const unsigned ldsCol = (unsigned)(size_t)(&xtile[0]) + (unsigned)tid * AGG_STRIDE;

  v8f acc[8];
#pragma unroll
  for (int i = 0; i < 8; ++i) acc[i] = (v8f){0.f, 0.f, 0.f, 0.f, 0.f, 0.f, 0.f, 0.f};

  for (int ks = 0; ks < 16; ++ks) {
    const int j0 = ks * 32;
    // ---- async stage: 64B per column (4 x b128), 8KB total per block ----
    const unsigned char* g = (const unsigned char*)(gcol + j0);
#pragma unroll
    for (int ch = 0; ch < 4; ++ch) {
      asm volatile("global_load_async_to_lds_b128 %0, %1, off"
                   :: "v"(ldsCol + ch * 16), "v"(g + ch * 16)
                   : "memory");
    }
    // overlap: A operand fetch from global while async DMA fills LDS
    const v16bf a = load2x16B(arow + j0 + koffA, arow + j0 + 16 + koffA);
    if (ks + 1 < 16) __builtin_prefetch(arow + (ks + 1) * 32, 0, 0);

    asm volatile("s_wait_asynccnt 0x0" ::: "memory");
    __syncthreads();

#pragma unroll
    for (int ct = 0; ct < 8; ++ct) {
      const unsigned char* lb = &xtile[(ct * 16 + rr) * AGG_STRIDE + koffB * 2];
      BfVec u;
      u.q[0] = *reinterpret_cast<const uint4*>(lb);
      u.q[1] = *reinterpret_cast<const uint4*>(lb + 32);
      acc[ct] = __builtin_amdgcn_wmma_f32_16x16x32_bf16(false, a, false, u.v, (short)0,
                                                        acc[ct], false, false);
    }
    __syncthreads();   // protect LDS panel before next stage overwrites it
  }

  const int rowBase = n0 + (hiHalf ? 8 : 0);
#pragma unroll
  for (int ct = 0; ct < 8; ++ct) {
    const int col = cg * AGG_COLS + ct * 16 + rr;  // col = c*T + t
    const int c = col >> 4;                        // T == 16
    const int t = col & 15;
    float* obase = out + (((size_t)b * CDIM + c) * NDIM) * TDIM + t;
#pragma unroll
    for (int r = 0; r < 8; ++r) obase[(size_t)(rowBase + r) * TDIM] = acc[ct][r];
  }
}

// ---------------------------------------------------------------------------
extern "C" void kernel_launch(void* const* d_in, const int* in_sizes, int n_in,
                              void* d_out, int out_size, void* d_ws, size_t ws_size,
                              hipStream_t stream) {
  const float* x          = (const float*)d_in[0];
  const float* Wq         = (const float*)d_in[1];
  const float* Wk         = (const float*)d_in[2];
  const float* structural = (const float*)d_in[3];
  const float* fusion_w   = (const float*)d_in[4];
  const float* fusion_b   = (const float*)d_in[5];
  float* out = (float*)d_out;

  char* ws = (char*)d_ws;
  const size_t MB = 1024u * 1024u;
  __hip_bfloat16* Qb        = (__hip_bfloat16*)(ws);             //  8 MB  [B,T,N,C]
  __hip_bfloat16* Kb        = (__hip_bfloat16*)(ws + 8  * MB);   //  8 MB  [B,T,N,C]
  __hip_bfloat16* xT        = (__hip_bfloat16*)(ws + 16 * MB);   //  8 MB  [B,C,T,N]
  float*          att_node  = (float*)         (ws + 24 * MB);   //  4 MB  [B,N,N]
  float*          att_struct= (float*)         (ws + 28 * MB);   //  1 MB  [N,N]
  __hip_bfloat16* fused_bf  = (__hip_bfloat16*)(ws + 29 * MB);   //  2 MB  [B,N,N]

  proj_qk_kernel<<<BDIM * TDIM * NDIM, 128, 0, stream>>>(x, Wq, Wk, Qb, Kb, xT);
  struct_softmax_kernel<<<NDIM, 256, 0, stream>>>(structural, att_struct);
  // 160KB dynamic LDS: 4 per-wave 32KB score scratchpads + 32KB shared accumulator
  scores_softmax_kernel<<<BDIM * (NDIM / 16), 128, 160 * 1024, stream>>>(Qb, Kb, att_node);
  fuse_softmax_kernel<<<BDIM * NDIM, 256, 0, stream>>>(att_node, att_struct,
                                                       fusion_w, fusion_b, fused_bf);
  aggregate_kernel<<<BDIM * 8 * 16, 128, 0, stream>>>(fused_bf, xT, out);
}
